// Decoder_63866163691780
// MI455X (gfx1250) — compile-verified, matrix operand order checked by reading, not run on hardware
//
#include <hip/hip_runtime.h>
#include <hip/hip_bf16.h>
#include <math.h>

typedef __attribute__((ext_vector_type(16))) _Float16 v16h;
typedef __attribute__((ext_vector_type(8)))  _Float16 v8h;
typedef __attribute__((ext_vector_type(4)))  _Float16 v4h;
typedef __attribute__((ext_vector_type(8)))  float    v8f;

// ---------------------------------------------------------------------------
// WMMA fragment loader (CDNA5 16-bit A/B layout, 16x32 K-major per lane):
// lane holds row/col (lane&15); half h = lane>>4; elements 0..7 = K[8h..8h+7],
// elements 8..15 = K[16+8h..23+8h]. Two ds_load_b128 per fragment.
// ---------------------------------------------------------------------------
__device__ __forceinline__ v16h load_frag(const _Float16* p, int h) {
  const v8h* q = (const v8h*)p;
  v8h lo = q[h];
  v8h hi = q[2 + h];
  v16h f;
#pragma unroll
  for (int i = 0; i < 8; ++i) { f[i] = lo[i]; f[i + 8] = hi[i]; }
  return f;
}

__device__ __forceinline__ v8f wmma_f16(v16h a, v16h b, v8f c) {
  return __builtin_amdgcn_wmma_f32_16x16x32_f16(false, a, false, b,
                                                (short)0, c, false, false);
}

__device__ __forceinline__ v4h cvt4(float4 f, float s) {
  v4h h;
  h[0] = (_Float16)(f.x * s); h[1] = (_Float16)(f.y * s);
  h[2] = (_Float16)(f.z * s); h[3] = (_Float16)(f.w * s);
  return h;
}

// ---------------------------------------------------------------------------
// GEMM: C[M,N] = A[M,K] * W[K,N] + bias[N], optional ReLU.
// 256 threads (8 waves), tile 128x128, K-step 32, double-buffered LDS.
// Each wave computes a 16x128 strip: 1 A-frag + 8 B-frags -> 8 WMMAs / K-step.
// Global loads for tile k+1 are issued before the WMMAs of tile k.
// ---------------------------------------------------------------------------
__global__ __launch_bounds__(256)
void gemm_bias_kernel(const float* __restrict__ A, const float* __restrict__ W,
                      const float* __restrict__ bias, float* __restrict__ C,
                      int M, int N, int K, int relu) {
  __shared__ __attribute__((aligned(16))) _Float16 As[2][128 * 40];
  __shared__ __attribute__((aligned(16))) _Float16 Bs[2][128 * 40];
  const int tid  = threadIdx.x;
  const int wave = tid >> 5, lane = tid & 31;
  const int h = lane >> 4, l16 = lane & 15;
  const int m0 = blockIdx.x * 128, n0 = blockIdx.y * 128;

  const int am  = tid >> 3;            // A: rows am + i*32, float4 col ac4
  const int ac4 = tid & 7;
  const int bn  = tid & 127;           // B: one column, 16 k-values
  const int bkg = (tid >> 7) << 4;     // k offset 0 or 16

  float4 ra[4];
  float  rb[16];

  auto issue = [&](int k0) {
#pragma unroll
    for (int i = 0; i < 4; ++i)
      ra[i] = *(const float4*)&A[(size_t)(m0 + am + i * 32) * K + k0 + ac4 * 4];
#pragma unroll
    for (int j = 0; j < 16; ++j)       // coalesced across lanes (consecutive n)
      rb[j] = W[(size_t)(k0 + bkg + j) * N + (n0 + bn)];
  };
  auto stash = [&](int buf) {
    _Float16* as = As[buf];
    _Float16* bs = Bs[buf];
#pragma unroll
    for (int i = 0; i < 4; ++i)
      *(v4h*)&as[(am + i * 32) * 40 + ac4 * 4] = cvt4(ra[i], 1.f);
    v8h b0, b1;
#pragma unroll
    for (int j = 0; j < 8; ++j) {
      b0[j] = (_Float16)rb[j];
      b1[j] = (_Float16)rb[j + 8];
    }
    *(v8h*)&bs[bn * 40 + bkg]     = b0;   // Bs is [n][k] (transposed)
    *(v8h*)&bs[bn * 40 + bkg + 8] = b1;
  };

  v8f acc[8] = {};
  issue(0);
  stash(0);
  __syncthreads();

  const int nk = K >> 5;
  for (int kt = 0; kt < nk; ++kt) {
    const int cur = kt & 1;
    if (kt + 1 < nk) issue((kt + 1) << 5);   // in flight during WMMAs

    v16h af = load_frag(&As[cur][(wave * 16 + l16) * 40], h);
#pragma unroll
    for (int nb = 0; nb < 8; ++nb) {
      v16h bf = load_frag(&Bs[cur][(nb * 16 + l16) * 40], h);
      acc[nb] = wmma_f16(af, bf, acc[nb]);
    }
    if (kt + 1 < nk) stash(cur ^ 1);
    __syncthreads();
  }

#pragma unroll
  for (int r = 0; r < 8; ++r) {
    int m = m0 + wave * 16 + r + 8 * h;      // C layout: M = elem + 8*half
#pragma unroll
    for (int nb = 0; nb < 8; ++nb) {
      int n = n0 + nb * 16 + l16;
      float v = acc[nb][r] + bias[n];
      if (relu) v = fmaxf(v, 0.f);
      C[(size_t)m * N + n] = v;
    }
  }
}

// ---------------------------------------------------------------------------
// Flash attention, WMMA f16, f32 accumulate. D=1024, H=16, DK=64.
// 128 threads; each wave owns a 16-row Q tile. Keys streamed in tiles of 32.
// ---------------------------------------------------------------------------
__global__ __launch_bounds__(128)
void attn_kernel(const float* __restrict__ Q, const float* __restrict__ K,
                 const float* __restrict__ V, float* __restrict__ O,
                 int Sq, int Sk, int causal) {
  __shared__ __attribute__((aligned(16))) _Float16 Qs[64 * 72];     // [m][dk]
  __shared__ __attribute__((aligned(16))) _Float16 Ks[32 * 72];     // [key][dk]
  __shared__ __attribute__((aligned(16))) _Float16 Vt[64 * 40];     // [dk][key]
  __shared__ __attribute__((aligned(16))) _Float16 Ps[4 * 16 * 40]; // per-wave P

  const int tid  = threadIdx.x;
  const int wave = tid >> 5, lane = tid & 31;
  const int h = lane >> 4, l16 = lane & 15;
  const int b = blockIdx.z, hd = blockIdx.y, q0 = blockIdx.x * 64;

  const float* Qb = Q + ((size_t)b * Sq) * 1024 + hd * 64;
  const float* Kb = K + ((size_t)b * Sk) * 1024 + hd * 64;
  const float* Vb = V + ((size_t)b * Sk) * 1024 + hd * 64;

  // Q tile 64x64, pre-scaled by 1/sqrt(DK): batched float4 loads, then store
  {
    float4 rq[8];
#pragma unroll
    for (int i = 0; i < 8; ++i) {
      int e = tid + i * 128;
      rq[i] = *(const float4*)&Qb[(size_t)(q0 + (e >> 4)) * 1024 + (e & 15) * 4];
    }
#pragma unroll
    for (int i = 0; i < 8; ++i) {
      int e = tid + i * 128;
      *(v4h*)&Qs[(e >> 4) * 72 + (e & 15) * 4] = cvt4(rq[i], 0.125f);
    }
  }

  const int kc4  = tid & 15;           // K tile: float4 column
  const int krow = tid >> 4;           // K tile: rows krow + i*8
  const int vd   = tid & 63;           // V tile: dk column
  const int vkg  = (tid >> 6) << 4;    // V tile: key offset 0 or 16

  float mi[8], li[8];
  v8f oa[4] = {};
#pragma unroll
  for (int r = 0; r < 8; ++r) { mi[r] = -INFINITY; li[r] = 0.f; }

  int kend = causal ? (q0 + 64) : Sk;
  if (kend > Sk) kend = Sk;

  for (int k0 = 0; k0 < kend; k0 += 32) {
    __syncthreads();
    {
      float4 rk[4];
      float  rv[16];
#pragma unroll
      for (int i = 0; i < 4; ++i)
        rk[i] = *(const float4*)&Kb[(size_t)(k0 + krow + i * 8) * 1024 + kc4 * 4];
#pragma unroll
      for (int j = 0; j < 16; ++j)     // coalesced across lanes (consecutive d)
        rv[j] = Vb[(size_t)(k0 + vkg + j) * 1024 + vd];
#pragma unroll
      for (int i = 0; i < 4; ++i)
        *(v4h*)&Ks[(krow + i * 8) * 72 + kc4 * 4] = cvt4(rk[i], 1.f);
      v8h v0h, v1h;
#pragma unroll
      for (int j = 0; j < 8; ++j) {
        v0h[j] = (_Float16)rv[j];
        v1h[j] = (_Float16)rv[j + 8];
      }
      *(v8h*)&Vt[vd * 40 + vkg]     = v0h;   // Vt is [dk][key]
      *(v8h*)&Vt[vd * 40 + vkg + 8] = v1h;
    }
    __syncthreads();

    // scores: Q(16x64) . K^T -> two 16x16 tiles (keys 0-15, 16-31)
    v8f s0 = {}, s1 = {};
#pragma unroll
    for (int c = 0; c < 2; ++c) {
      v16h qf  = load_frag(&Qs[(wave * 16 + l16) * 72 + c * 32], h);
      v16h k0f = load_frag(&Ks[l16 * 72 + c * 32], h);
      v16h k1f = load_frag(&Ks[(16 + l16) * 72 + c * 32], h);
      s0 = wmma_f16(qf, k0f, s0);
      s1 = wmma_f16(qf, k1f, s1);
    }

    _Float16* Pw = &Ps[wave * 16 * 40];
#pragma unroll
    for (int r = 0; r < 8; ++r) {
      int mloc = r + 8 * h;
      int qg = q0 + wave * 16 + mloc;
      int kg0 = k0 + l16, kg1 = k0 + 16 + l16;
      float v0 = s0[r], v1 = s1[r];
      // reference semantics: exact-zero scores are "padding" -> masked
      if (v0 == 0.f || (causal && kg0 > qg)) v0 = -1e9f;
      if (v1 == 0.f || (causal && kg1 > qg)) v1 = -1e9f;
      float cm = fmaxf(v0, v1);
      cm = fmaxf(cm, __shfl_xor(cm, 1));
      cm = fmaxf(cm, __shfl_xor(cm, 2));
      cm = fmaxf(cm, __shfl_xor(cm, 4));
      cm = fmaxf(cm, __shfl_xor(cm, 8));
      float mnew = fmaxf(mi[r], cm);
      float sc = __expf(mi[r] - mnew);
      float p0 = __expf(v0 - mnew), p1 = __expf(v1 - mnew);
      float rs = p0 + p1;
      rs += __shfl_xor(rs, 1);
      rs += __shfl_xor(rs, 2);
      rs += __shfl_xor(rs, 4);
      rs += __shfl_xor(rs, 8);
      li[r] = li[r] * sc + rs;
      mi[r] = mnew;
#pragma unroll
      for (int nb = 0; nb < 4; ++nb) oa[nb][r] *= sc;
      Pw[mloc * 40 + l16]      = (_Float16)p0;   // C-layout -> A-layout via LDS
      Pw[mloc * 40 + 16 + l16] = (_Float16)p1;
    }

    // O += P(16x32) . V(32x64)
    v16h pf = load_frag(&Pw[l16 * 40], h);
#pragma unroll
    for (int nb = 0; nb < 4; ++nb) {
      v16h vf = load_frag(&Vt[(nb * 16 + l16) * 40], h);
      oa[nb] = wmma_f16(pf, vf, oa[nb]);
    }
  }

  float* Ob = O + ((size_t)b * Sq) * 1024 + hd * 64;
#pragma unroll
  for (int r = 0; r < 8; ++r) {
    int m = q0 + wave * 16 + r + 8 * h;
    float inv = li[r] > 0.f ? 1.f / li[r] : 0.f;
#pragma unroll
    for (int nb = 0; nb < 4; ++nb)
      Ob[(size_t)m * 1024 + nb * 16 + l16] = oa[nb][r] * inv;
  }
}

// ---------------------------------------------------------------------------
// out[row] = LayerNorm(a[row] + b[row]) * g + beta.  One block per row, D=1024.
// ---------------------------------------------------------------------------
__global__ __launch_bounds__(256)
void ln_residual_kernel(const float* __restrict__ a, const float* __restrict__ b,
                        const float* __restrict__ g, const float* __restrict__ be,
                        float* __restrict__ out) {
  __shared__ float red[256];
  const int row = blockIdx.x, tid = threadIdx.x;
  const size_t base = (size_t)row * 1024;
  const int c = tid * 4;

  float4 va = *(const float4*)&a[base + c];
  float4 vb = *(const float4*)&b[base + c];
  float v[4] = { va.x + vb.x, va.y + vb.y, va.z + vb.z, va.w + vb.w };
  float s = v[0] + v[1] + v[2] + v[3];

  red[tid] = s; __syncthreads();
  for (int off = 128; off > 0; off >>= 1) {
    if (tid < off) red[tid] += red[tid + off];
    __syncthreads();
  }
  float mean = red[0] * (1.f / 1024.f);
  __syncthreads();

  float s2 = 0.f;
#pragma unroll
  for (int i = 0; i < 4; ++i) { float d = v[i] - mean; s2 += d * d; }
  red[tid] = s2; __syncthreads();
  for (int off = 128; off > 0; off >>= 1) {
    if (tid < off) red[tid] += red[tid + off];
    __syncthreads();
  }
  float inv = rsqrtf(red[0] * (1.f / 1024.f) + 1e-5f);

  float4 vg  = *(const float4*)&g[c];
  float4 vbe = *(const float4*)&be[c];
  float4 o;
  o.x = (v[0] - mean) * inv * vg.x + vbe.x;
  o.y = (v[1] - mean) * inv * vg.y + vbe.y;
  o.z = (v[2] - mean) * inv * vg.z + vbe.z;
  o.w = (v[3] - mean) * inv * vg.w + vbe.w;
  *(float4*)&out[base + c] = o;
}

// ---------------------------------------------------------------------------
// Host orchestration: 6 decoder blocks.
// ---------------------------------------------------------------------------
extern "C" void kernel_launch(void* const* d_in, const int* in_sizes, int n_in,
                              void* d_out, int out_size, void* d_ws, size_t ws_size,
                              hipStream_t stream) {
  const int D = 1024, DF = 4096, Lb = 6;
  const int M = 4096;                 // B*S tokens
  const size_t SZ = (size_t)M * D;

  const float* enc  = (const float*)d_in[0];
  const float* x_in = (const float*)d_in[1];
  const float* wq1 = (const float*)d_in[2];  const float* bq1 = (const float*)d_in[3];
  const float* wk1 = (const float*)d_in[4];  const float* bk1 = (const float*)d_in[5];
  const float* wv1 = (const float*)d_in[6];  const float* bv1 = (const float*)d_in[7];
  const float* wq2 = (const float*)d_in[8];  const float* bq2 = (const float*)d_in[9];
  const float* wk2 = (const float*)d_in[10]; const float* bk2 = (const float*)d_in[11];
  const float* wv2 = (const float*)d_in[12]; const float* bv2 = (const float*)d_in[13];
  const float* g1  = (const float*)d_in[14]; const float* be1 = (const float*)d_in[15];
  const float* g2  = (const float*)d_in[16]; const float* be2 = (const float*)d_in[17];
  const float* g3  = (const float*)d_in[18]; const float* be3 = (const float*)d_in[19];
  const float* wf1 = (const float*)d_in[20]; const float* bf1 = (const float*)d_in[21];
  const float* wf2 = (const float*)d_in[22]; const float* bf2 = (const float*)d_in[23];

  float* X  = (float*)d_ws;
  float* Qb = X  + SZ;
  float* Kb = Qb + SZ;
  float* Vb = Kb + SZ;
  float* Ab = Vb + SZ;
  float* Hb = Ab + SZ;                // 4096 x 4096
  size_t need = (5 * SZ + (size_t)M * DF) * sizeof(float);
  if (ws_size < need) return;

  hipMemcpyAsync(X, x_in, SZ * sizeof(float), hipMemcpyDeviceToDevice, stream);

  dim3 gemmblk(256);
  dim3 gproj(M / 128, D / 128);       // 32 x 8
  dim3 gff1 (M / 128, DF / 128);      // 32 x 32
  dim3 gff2 (M / 128, D / 128);
  dim3 gattn(1024 / 64, 16, 4);       // q-tiles x heads x batch

  for (int i = 0; i < Lb; ++i) {
    size_t wo = (size_t)i * D * D, bo = (size_t)i * D;
    size_t wf1o = (size_t)i * D * DF, bf1o = (size_t)i * DF;
    size_t wf2o = (size_t)i * DF * D;

    // ---- masked self-attention ----
    gemm_bias_kernel<<<gproj, gemmblk, 0, stream>>>(X, wq1 + wo, bq1 + bo, Qb, M, D, D, 0);
    gemm_bias_kernel<<<gproj, gemmblk, 0, stream>>>(X, wk1 + wo, bk1 + bo, Kb, M, D, D, 0);
    gemm_bias_kernel<<<gproj, gemmblk, 0, stream>>>(X, wv1 + wo, bv1 + bo, Vb, M, D, D, 0);
    attn_kernel<<<gattn, dim3(128), 0, stream>>>(Qb, Kb, Vb, Ab, 1024, 1024, 1);
    ln_residual_kernel<<<M, 256, 0, stream>>>(Ab, X, g1 + bo, be1 + bo, X);

    // ---- cross-attention ----
    gemm_bias_kernel<<<gproj, gemmblk, 0, stream>>>(X,   wq2 + wo, bq2 + bo, Qb, M, D, D, 0);
    gemm_bias_kernel<<<gproj, gemmblk, 0, stream>>>(enc, wk2 + wo, bk2 + bo, Kb, M, D, D, 0);
    gemm_bias_kernel<<<gproj, gemmblk, 0, stream>>>(enc, wv2 + wo, bv2 + bo, Vb, M, D, D, 0);
    attn_kernel<<<gattn, dim3(128), 0, stream>>>(Qb, Kb, Vb, Ab, 1024, 1024, 0);
    ln_residual_kernel<<<M, 256, 0, stream>>>(Ab, X, g2 + bo, be2 + bo, X);

    // ---- FFN ----
    gemm_bias_kernel<<<gff1, gemmblk, 0, stream>>>(X,  wf1 + wf1o, bf1 + bf1o, Hb, M, DF, D, 1);
    gemm_bias_kernel<<<gff2, gemmblk, 0, stream>>>(Hb, wf2 + wf2o, bf2 + bo,   Ab, M, D, DF, 0);
    ln_residual_kernel<<<M, 256, 0, stream>>>(Ab, X, g3 + bo, be3 + bo, X);
  }

  hipMemcpyAsync(d_out, X, (size_t)out_size * sizeof(float),
                 hipMemcpyDeviceToDevice, stream);
}
